// Model_60524679135666
// MI455X (gfx1250) — compile-verified, matrix operand order checked by reading, not run on hardware
//
#include <hip/hip_runtime.h>
#include <hip/hip_bf16.h>

// ---------------------------------------------------------------------------
// MI455X (gfx1250) implementation: graph-attention block, f16 WMMA everywhere.
// ---------------------------------------------------------------------------

typedef _Float16 half_t;
typedef __attribute__((ext_vector_type(16))) _Float16 v16h;
typedef __attribute__((ext_vector_type(8)))  float    v8f;

#define NB   4
#define CIN  64
#define NN   256
#define NP   257
#define TT   96
#define LP   96
#define COLS (NP * LP)     // 24672 = 257*96, multiple of 16
#define NT16 1542          // COLS/16
#define BN_CNT 98688       // NB*NP*TT per channel

// f16 A/B fragment K-pair base for 16x16x32 WMMA (ISA 7.12.2):
// VGPR g holds K pair {base, base+1}; lanes16-31 offset K by +8.
__device__ __forceinline__ int kbase(int g, int h) {
  return ((g < 4) ? (2 * g) : (16 + 2 * (g - 4))) + 8 * h;
}

__device__ __forceinline__ v8f wmma16(v16h a, v16h b, v8f c) {
  return __builtin_amdgcn_wmma_f32_16x16x32_f16(false, a, false, b, (short)0, c,
                                                false, false);
}

// DS ops are in-order within a wave; this orders LDS write->read across lanes
// of the same wave and stops compiler reordering.
#define LDS_SYNC() asm volatile("s_wait_dscnt 0" ::: "memory")

// ---------------------------------------------------------------------------
// Utility kernels
// ---------------------------------------------------------------------------
__global__ void zero_f32_kernel(float* p, long n) {
  long i = (long)blockIdx.x * blockDim.x + threadIdx.x;
  if (i < n) p[i] = 0.f;
}
__global__ void zero_f16_kernel(half_t* p, long n) {
  long i = (long)blockIdx.x * blockDim.x + threadIdx.x;
  if (i < n) p[i] = (half_t)0;
}
__global__ void pack_f16_kernel(const float* __restrict__ s, half_t* __restrict__ d, long n) {
  long i = (long)blockIdx.x * blockDim.x + threadIdx.x;
  if (i < n) d[i] = (half_t)s[i];
}
// fo_w (64 x L) -> (64 x 96) f16, zero-padded tail
__global__ void pack_fow_kernel(const float* __restrict__ s, half_t* __restrict__ d, int L) {
  int i = blockIdx.x * blockDim.x + threadIdx.x;
  if (i >= 64 * 96) return;
  int o = i / 96, l = i % 96;
  d[i] = (l < L) ? (half_t)s[o * L + l] : (half_t)0;
}
// [fo_w(64x256) | fr_w(64x64)] -> (64 x 320) f16
__global__ void pack_finalA_kernel(const float* __restrict__ fo, const float* __restrict__ fr,
                                   half_t* __restrict__ d) {
  int i = blockIdx.x * blockDim.x + threadIdx.x;
  if (i >= 64 * 320) return;
  int o = i / 320, k = i % 320;
  d[i] = (half_t)((k < 256) ? fo[o * 256 + k] : fr[o * 64 + (k - 256)]);
}

// ---------------------------------------------------------------------------
// Embedding fusion: xe[b,c,n,t] = pad(x) + time + day + node + degree embs.
// ---------------------------------------------------------------------------
__global__ void embed_kernel(const float* __restrict__ x, const int* __restrict__ idx,
                             const int* __restrict__ node_ids, const int* __restrict__ degrees,
                             const float* __restrict__ time_emb, const float* __restrict__ day_emb,
                             const float* __restrict__ node_emb, const float* __restrict__ degree_emb,
                             float* __restrict__ xe, half_t* __restrict__ xe_h) {
  long gid = (long)blockIdx.x * blockDim.x + threadIdx.x;
  if (gid >= (long)NB * CIN * NP * TT) return;
  int t = (int)(gid % TT);
  int n = (int)((gid / TT) % NP);
  int c = (int)((gid / ((long)TT * NP)) % CIN);
  int b = (int)(gid / ((long)TT * NP * CIN));
  float v = (n < NN) ? x[(((long)b * CIN + c) * NN + n) * TT + t] : 0.f;
  int ti = idx[(b * 2 + 0) * TT + t];
  int di = idx[(b * 2 + 1) * TT + t];
  v += time_emb[ti * CIN + c];
  v += day_emb[di * CIN + c];
  v += node_emb[node_ids[n] * CIN + c];
  v += degree_emb[degrees[n] * CIN + c];
  xe[gid]   = v;
  xe_h[gid] = (half_t)v;
}

// ---------------------------------------------------------------------------
// Time-conv as implicit-im2col WMMA GEMM:
//   y[b,to,(n,l)] = sum_{c,dt} w[to,c,dt] * xe[b,c,n,l+dt] + bias[to]
// M=96 (6 tiles), N=COLS (1542 tiles), K=64*TF.  Output f16, l>=L zero-padded.
// One wave per 16x16 tile; grid exact => EXEC always full.
// ---------------------------------------------------------------------------
template <int TF>
__global__ __launch_bounds__(256) void conv_gemm_kernel(
    const half_t* __restrict__ xe_h, const half_t* __restrict__ w_h,
    const float* __restrict__ bias, half_t* __restrict__ y, int L) {
  constexpr int K = 64 * TF;
  int wid = (blockIdx.x * 256 + threadIdx.x) >> 5;
  int lane = threadIdx.x & 31, l16 = lane & 15, h = lane >> 4;
  int nt = wid % NT16;
  int rem = wid / NT16;
  int mt = rem % 6;
  int b  = rem / 6;
  int col = nt * 16 + l16;
  int n = col / LP, l = col % LP;
  int arow = mt * 16 + l16;
  const half_t* xb = xe_h + (long)b * CIN * NP * TT;
  v8f acc = {};
  for (int kk = 0; kk < K; kk += 32) {
    v16h a, bf;
#pragma unroll
    for (int g = 0; g < 8; ++g) {
      int kb = kk + kbase(g, h);
      a[2 * g]     = w_h[arow * K + kb];
      a[2 * g + 1] = w_h[arow * K + kb + 1];
#pragma unroll
      for (int e = 0; e < 2; ++e) {
        int k = kb + e;
        int c = k / TF, dt = k - c * TF;
        half_t v = (half_t)0;
        if (l < L) v = xb[((long)c * NP + n) * TT + (l + dt)];
        bf[2 * g + e] = v;
      }
    }
    acc = wmma16(a, bf, acc);
  }
#pragma unroll
  for (int r = 0; r < 8; ++r) {   // D: row = r+8*h, col = l16
    int to = mt * 16 + r + 8 * h;
    float v = acc[r] + bias[to];
    if (l >= L) v = 0.f;          // zero-pad l-tail so downstream K-dims are clean
    y[((long)b * 96 + to) * COLS + col] = (half_t)v;
  }
}

// ---------------------------------------------------------------------------
// Fold fo_w into V:  vpT[b,t,o,m] = sum_l v[b,t,m,l] * fo_w[o,l]
// M = m (17 tiles, rows>=257 -> 0), N = o (4 tiles), K = 96.
// Output transposed (o-major, m padded to 288) for dword-friendly B frags.
// ---------------------------------------------------------------------------
__global__ __launch_bounds__(256) void vp_gemm_kernel(
    const half_t* __restrict__ v_h, const half_t* __restrict__ fow,
    half_t* __restrict__ vpT) {
  int wid = (blockIdx.x * 256 + threadIdx.x) >> 5;
  int lane = threadIdx.x & 31, l16 = lane & 15, h = lane >> 4;
  int ot = wid & 3;
  int mt = (wid >> 2) % 17;
  int bt = wid / (4 * 17);
  const half_t* vr = v_h + (long)bt * NP * LP;
  int m = mt * 16 + l16; if (m > 256) m = 256;
  int o = ot * 16 + l16;
  v8f acc = {};
  for (int kk = 0; kk < 96; kk += 32) {
    v16h a, bf;
#pragma unroll
    for (int g = 0; g < 8; ++g) {
      int kb = kk + kbase(g, h);
      a[2 * g]      = vr[m * LP + kb];
      a[2 * g + 1]  = vr[m * LP + kb + 1];
      bf[2 * g]     = fow[o * 96 + kb];
      bf[2 * g + 1] = fow[o * 96 + kb + 1];
    }
    acc = wmma16(a, bf, acc);
  }
#pragma unroll
  for (int r = 0; r < 8; ++r) {
    int mo = mt * 16 + r + 8 * h;
    int oc = ot * 16 + l16;
    float v = (mo < NP) ? acc[r] : 0.f;
    vpT[((long)bt * 64 + oc) * 288 + mo] = (half_t)v;
  }
}

// ---------------------------------------------------------------------------
// Per-head residual base:  hout[b,o,(n,t)] = fr_w @ xe + fr_b + fo_b
// M=64 (4 tiles), N=COLS, K=64.
// ---------------------------------------------------------------------------
__global__ __launch_bounds__(256) void res_gemm_kernel(
    const half_t* __restrict__ xe_h, const half_t* __restrict__ frw_h,
    const float* __restrict__ frb, const float* __restrict__ fob,
    float* __restrict__ hout) {
  int wid = (blockIdx.x * 256 + threadIdx.x) >> 5;
  int lane = threadIdx.x & 31, l16 = lane & 15, h = lane >> 4;
  int nt = wid % NT16;
  int rem = wid / NT16;
  int mt = rem % 4;
  int b  = rem / 4;
  int col = nt * 16 + l16;
  int orow = mt * 16 + l16;
  const half_t* xb = xe_h + (long)b * CIN * COLS;
  v8f acc = {};
  for (int kk = 0; kk < 64; kk += 32) {
    v16h a, bf;
#pragma unroll
    for (int g = 0; g < 8; ++g) {
      int kb = kk + kbase(g, h);
      a[2 * g]      = frw_h[orow * 64 + kb];
      a[2 * g + 1]  = frw_h[orow * 64 + kb + 1];
      bf[2 * g]     = xb[(long)kb * COLS + col];
      bf[2 * g + 1] = xb[(long)(kb + 1) * COLS + col];
    }
    acc = wmma16(a, bf, acc);
  }
#pragma unroll
  for (int r = 0; r < 8; ++r) {
    int o = mt * 16 + r + 8 * h;
    hout[((long)b * 64 + o) * COLS + col] = acc[r] + frb[o] + fob[o];
  }
}

// ---------------------------------------------------------------------------
// Fused attention: per (b,t) workgroup.  S = K·Qᵀ (+adj+adj_hat, padded row/col
// = 1), flash-style online softmax over m, O += P · VPᵀ, all in WMMA; the
// 257x257 score never leaves the WGP.  8 waves, each owns independent row tiles
// (no cross-wave sync; intra-wave LDS ordering via s_wait_dscnt).
// K-matrix A-fragments are hoisted out of the column loop (reused 17x).
// ---------------------------------------------------------------------------
__global__ __launch_bounds__(256) void attn_kernel(
    const half_t* __restrict__ k_h, const half_t* __restrict__ q_h,
    const half_t* __restrict__ vpT, const float* __restrict__ adj,
    const float* __restrict__ adjh, float* __restrict__ hout) {
  __shared__ float lds[8 * 304];
  int b = blockIdx.x / TT, t = blockIdx.x % TT;
  int wave = threadIdx.x >> 5, lane = threadIdx.x & 31;
  int l16 = lane & 15, h = lane >> 4;
  float* Sw     = lds + wave * 304;
  float* rmax   = Sw + 256;
  float* rsum   = Sw + 272;
  float* rscale = Sw + 288;
  long bt = (long)b * TT + t;
  const half_t* Kp  = k_h + bt * NP * LP;
  const half_t* Qp  = q_h + bt * NP * LP;
  const half_t* VPp = vpT + bt * 64 * 288;

  for (int ntile = wave; ntile < 17; ntile += 8) {   // wave-uniform
    if (lane < 16) { rmax[lane] = -1e30f; rsum[lane] = 0.f; }
    LDS_SYNC();
    v8f oacc[4];
    for (int i = 0; i < 4; ++i) oacc[i] = v8f{};
    int nrow = ntile * 16 + l16; if (nrow > 256) nrow = 256;

    // Hoisted K-row A-fragments: constant across all 17 column tiles.
    v16h kfrag[3];
#pragma unroll
    for (int kkI = 0; kkI < 3; ++kkI) {
      int kk = kkI * 32;
#pragma unroll
      for (int g = 0; g < 8; ++g) {
        int kb = kk + kbase(g, h);
        kfrag[kkI][2 * g]     = Kp[nrow * LP + kb];
        kfrag[kkI][2 * g + 1] = Kp[nrow * LP + kb + 1];
      }
    }

    for (int jt = 0; jt < 17; ++jt) {
      // ---- S tile: rows n (K-mat), cols m (Q-mat), K-dim 96 ----
      int mcol = jt * 16 + l16; if (mcol > 256) mcol = 256;
      __builtin_prefetch(VPp + (long)l16 * 288 + jt * 16, 0, 1);
      v8f sacc = {};
#pragma unroll
      for (int kkI = 0; kkI < 3; ++kkI) {
        int kk = kkI * 32;
        v16h bf;
#pragma unroll
        for (int g = 0; g < 8; ++g) {
          int kb = kk + kbase(g, h);
          bf[2 * g]     = Qp[mcol * LP + kb];
          bf[2 * g + 1] = Qp[mcol * LP + kb + 1];
        }
        sacc = wmma16(kfrag[kkI], bf, sacc);
      }
      // ---- stage to LDS with adjacency, mask invalid m ----
#pragma unroll
      for (int r = 0; r < 8; ++r) {
        int nl = r + 8 * h;
        int ng = ntile * 16 + nl;
        int mg = jt * 16 + l16;
        float sv = sacc[r];
        if (mg < NP) {
          float a1 = 1.f, a2 = 1.f;     // padded row/col value
          if (ng < NN && mg < NN) {
            a1 = adj [((long)b * NN + ng) * NN + mg];
            a2 = adjh[((long)b * NN + ng) * NN + mg];
          }
          sv += a1 + a2;
        } else {
          sv = -1e30f;                  // excluded from softmax
        }
        Sw[nl * 16 + l16] = sv;
      }
      LDS_SYNC();
      // ---- online softmax stats (lanes 0..15 own one row each) ----
      if (lane < 16) {
        float mloc = -1e30f;
#pragma unroll
        for (int j = 0; j < 16; ++j) mloc = fmaxf(mloc, Sw[lane * 16 + j]);
        float mold = rmax[lane];
        float mnew = fmaxf(mold, mloc);
        float sc = __expf(mold - mnew);
        float s = 0.f;
#pragma unroll
        for (int j = 0; j < 16; ++j) {
          float p = __expf(Sw[lane * 16 + j] - mnew);
          Sw[lane * 16 + j] = p;
          s += p;
        }
        rsum[lane] = rsum[lane] * sc + s;
        rmax[lane] = mnew;
        rscale[lane] = sc;
      }
      LDS_SYNC();
      // ---- O rescale + O += P @ VP (P as 16x32 A frag, K 16..31 zero) ----
      v16h pf;
#pragma unroll
      for (int g = 0; g < 8; ++g) {
        int kb = kbase(g, h);
        pf[2 * g]     = (kb     < 16) ? (half_t)Sw[l16 * 16 + kb]     : (half_t)0;
        pf[2 * g + 1] = (kb + 1 < 16) ? (half_t)Sw[l16 * 16 + kb + 1] : (half_t)0;
      }
      float scr[8];
#pragma unroll
      for (int r = 0; r < 8; ++r) scr[r] = rscale[r + 8 * h];
#pragma unroll
      for (int ot = 0; ot < 4; ++ot) {
#pragma unroll
        for (int r = 0; r < 8; ++r) oacc[ot][r] *= scr[r];
        v16h bf;
#pragma unroll
        for (int g = 0; g < 8; ++g) {
          int kb = kbase(g, h);
          int m0 = jt * 16 + kb;          // <= 287, vpT padded to 288
          int oc = ot * 16 + l16;
          bf[2 * g]     = VPp[(long)oc * 288 + m0];
          bf[2 * g + 1] = VPp[(long)oc * 288 + m0 + 1];
        }
        oacc[ot] = wmma16(pf, bf, oacc[ot]);
      }
    } // jt
    LDS_SYNC();
    float inv[8];
#pragma unroll
    for (int r = 0; r < 8; ++r) inv[r] = 1.f / rsum[r + 8 * h];
#pragma unroll
    for (int r = 0; r < 8; ++r) {
      int ng = ntile * 16 + r + 8 * h;
      if (ng < NP) {
#pragma unroll
        for (int ot = 0; ot < 4; ++ot) {
          int o = ot * 16 + l16;
          long oi = (((long)b * 64 + o) * NP + ng) * TT + t;
          hout[oi] = hout[oi] + oacc[ot][r] * inv[r];   // unique writer
        }
      }
    }
  }
}

// ---------------------------------------------------------------------------
// BatchNorm (training-mode) stats + apply
// ---------------------------------------------------------------------------
__global__ void bn_stats_kernel(const float* __restrict__ buf, float* __restrict__ stats) {
  __shared__ float s1[256], s2[256];
  int o = blockIdx.y;
  int i = blockIdx.x * 256 + threadIdx.x;
  float v = 0.f;
  if (i < BN_CNT) {
    int t = i % TT;
    int n = (i / TT) % NP;
    int b = i / (TT * NP);
    v = buf[(((long)b * 64 + o) * NP + n) * TT + t];
  }
  s1[threadIdx.x] = v;
  s2[threadIdx.x] = v * v;
  __syncthreads();
  for (int s = 128; s > 0; s >>= 1) {
    if (threadIdx.x < s) {
      s1[threadIdx.x] += s1[threadIdx.x + s];
      s2[threadIdx.x] += s2[threadIdx.x + s];
    }
    __syncthreads();
  }
  if (threadIdx.x == 0) {
    atomicAdd(&stats[o], s1[0]);
    atomicAdd(&stats[64 + o], s2[0]);
  }
}

__global__ void bn_apply_concat_kernel(const float* __restrict__ buf,
                                       const float* __restrict__ stats,
                                       const float* __restrict__ g,
                                       const float* __restrict__ be,
                                       half_t* __restrict__ concat, int head) {
  long gid = (long)blockIdx.x * blockDim.x + threadIdx.x;
  if (gid >= (long)NB * 64 * NP * TT) return;
  int t = (int)(gid % TT);
  int n = (int)((gid / TT) % NP);
  int o = (int)((gid / ((long)TT * NP)) % 64);
  int b = (int)(gid / ((long)TT * NP * 64));
  float mean = stats[o] * (1.f / (float)BN_CNT);
  float var  = stats[64 + o] * (1.f / (float)BN_CNT) - mean * mean;
  float w = rsqrtf(var + 1e-5f) * g[o];
  float val = (buf[gid] - mean) * w + be[o];
  concat[(((long)b * 256 + head * 64 + o) * NP + n) * TT + t] = (half_t)val;
}

// ---------------------------------------------------------------------------
// Final fused 1x1: out_pre = [fo_w | fr_w](64x320) @ [concat ; xe] + biases
// ---------------------------------------------------------------------------
__global__ __launch_bounds__(256) void final_gemm_kernel(
    const half_t* __restrict__ concat, const half_t* __restrict__ xe_h,
    const half_t* __restrict__ packA, const float* __restrict__ fob,
    const float* __restrict__ frb, float* __restrict__ fout) {
  int wid = (blockIdx.x * 256 + threadIdx.x) >> 5;
  int lane = threadIdx.x & 31, l16 = lane & 15, h = lane >> 4;
  int nt = wid % NT16;
  int rem = wid / NT16;
  int mt = rem % 4;
  int b  = rem / 4;
  int col = nt * 16 + l16;
  int orow = mt * 16 + l16;
  const half_t* cb = concat + (long)b * 256 * COLS;
  const half_t* xb = xe_h   + (long)b * CIN * COLS;
  v8f acc = {};
  for (int kk = 0; kk < 320; kk += 32) {
    v16h a, bf;
#pragma unroll
    for (int g = 0; g < 8; ++g) {
      int kb = kk + kbase(g, h);
      a[2 * g]     = packA[orow * 320 + kb];
      a[2 * g + 1] = packA[orow * 320 + kb + 1];
#pragma unroll
      for (int e = 0; e < 2; ++e) {
        int k = kb + e;
        bf[2 * g + e] = (k < 256) ? cb[(long)k * COLS + col]
                                  : xb[(long)(k - 256) * COLS + col];
      }
    }
    acc = wmma16(a, bf, acc);
  }
#pragma unroll
  for (int r = 0; r < 8; ++r) {
    int o = mt * 16 + r + 8 * h;
    fout[((long)b * 64 + o) * COLS + col] = acc[r] + fob[o] + frb[o];
  }
}

__global__ void final_out_kernel(const float* __restrict__ fout,
                                 const float* __restrict__ stats,
                                 const float* __restrict__ g,
                                 const float* __restrict__ be,
                                 float* __restrict__ out) {
  long gid = (long)blockIdx.x * blockDim.x + threadIdx.x;
  if (gid >= (long)NB * 64 * NN * TT) return;
  int t = (int)(gid % TT);
  int n = (int)((gid / TT) % NN);
  int o = (int)((gid / ((long)TT * NN)) % 64);
  int b = (int)(gid / ((long)TT * NN * 64));
  float mean = stats[o] * (1.f / (float)BN_CNT);
  float var  = stats[64 + o] * (1.f / (float)BN_CNT) - mean * mean;
  float w = rsqrtf(var + 1e-5f) * g[o];
  float v = fout[(((long)b * 64 + o) * NP + n) * TT + t];
  out[gid] = (v - mean) * w + be[o];
}

// ---------------------------------------------------------------------------
// Host orchestration
// ---------------------------------------------------------------------------
extern "C" void kernel_launch(void* const* d_in, const int* in_sizes, int n_in,
                              void* d_out, int out_size, void* d_ws, size_t ws_size,
                              hipStream_t stream) {
  (void)in_sizes; (void)n_in; (void)out_size; (void)ws_size;
  const float* x          = (const float*)d_in[0];
  const int*   idx        = (const int*)d_in[1];
  const float* adj        = (const float*)d_in[2];
  const float* adjh       = (const float*)d_in[3];
  const int*   node_ids   = (const int*)d_in[4];
  const int*   degrees    = (const int*)d_in[5];
  const float* time_emb   = (const float*)d_in[6];
  const float* day_emb    = (const float*)d_in[7];
  const float* node_emb   = (const float*)d_in[8];
  const float* degree_emb = (const float*)d_in[9];
  // mhga tail params
  const float* mfo_w = (const float*)d_in[58];
  const float* mfo_b = (const float*)d_in[59];
  const float* mfr_w = (const float*)d_in[60];
  const float* mfr_b = (const float*)d_in[61];
  const float* mbn_g = (const float*)d_in[62];
  const float* mbn_b = (const float*)d_in[63];

  char* ws = (char*)d_ws;
  float*  xe     = (float*)(ws + 0);                        // 25,264,128 B
  half_t* xe_h   = (half_t*)(ws + 25264128);                // 12,632,064 B
  half_t* buf0   = (half_t*)(ws + 37896192);                // 18,948,096 B (V then K)
  half_t* buf1   = (half_t*)(ws + 56844288);                // 18,948,096 B (Q)
  half_t* vpT    = (half_t*)(ws + 75792384);                // 14,155,776 B
  float*  hout   = (float*)(ws + 89948160);                 // 25,264,128 B
  half_t* concat = (half_t*)(ws + 115212288);               // 50,528,256 B
  half_t* wtmp   = (half_t*)(ws + 165740544);               // 86,016 B
  half_t* fow_h  = (half_t*)(ws + 165826560);               // 12,288 B
  half_t* packA  = (half_t*)(ws + 165838848);               // 40,960 B
  float*  stats  = (float*)(ws + 165879808);                // 512 B

  const long n_xe = (long)NB * CIN * NP * TT;               // 6,316,032
  const long n_vp = (long)NB * TT * 64 * 288;               // 7,077,888

  embed_kernel<<<(unsigned)((n_xe + 255) / 256), 256, 0, stream>>>(
      x, idx, node_ids, degrees, time_emb, day_emb, node_emb, degree_emb, xe, xe_h);

  const int tfs[4] = {2, 3, 6, 7};
  for (int hi = 0; hi < 4; ++hi) {
    int tf = tfs[hi];
    int L = TT - tf + 1;
    int pbase = 10 + hi * 12;
    const float* kw  = (const float*)d_in[pbase + 0];
    const float* kb  = (const float*)d_in[pbase + 1];
    const float* qw  = (const float*)d_in[pbase + 2];
    const float* qb  = (const float*)d_in[pbase + 3];
    const float* vw  = (const float*)d_in[pbase + 4];
    const float* vb  = (const float*)d_in[pbase + 5];
    const float* fow = (const float*)d_in[pbase + 6];
    const float* fob = (const float*)d_in[pbase + 7];
    const float* frw = (const float*)d_in[pbase + 8];
    const float* frb = (const float*)d_in[pbase + 9];
    const float* bng = (const float*)d_in[pbase + 10];
    const float* bnb = (const float*)d_in[pbase + 11];
    long nW = 96L * 64 * tf;

    // V conv -> buf0, fold fo_w -> vpT
    pack_f16_kernel<<<(unsigned)((nW + 255) / 256), 256, 0, stream>>>(vw, wtmp, nW);
    switch (tf) {
      case 2: conv_gemm_kernel<2><<<4626, 256, 0, stream>>>(xe_h, wtmp, vb, buf0, L); break;
      case 3: conv_gemm_kernel<3><<<4626, 256, 0, stream>>>(xe_h, wtmp, vb, buf0, L); break;
      case 6: conv_gemm_kernel<6><<<4626, 256, 0, stream>>>(xe_h, wtmp, vb, buf0, L); break;
      default: conv_gemm_kernel<7><<<4626, 256, 0, stream>>>(xe_h, wtmp, vb, buf0, L); break;
    }
    pack_fow_kernel<<<(64 * 96 + 255) / 256, 256, 0, stream>>>(fow, fow_h, L);
    zero_f16_kernel<<<(unsigned)((n_vp + 255) / 256), 256, 0, stream>>>(vpT, n_vp);
    vp_gemm_kernel<<<3264, 256, 0, stream>>>(buf0, fow_h, vpT);

    // K conv -> buf0 (reuse), Q conv -> buf1
    pack_f16_kernel<<<(unsigned)((nW + 255) / 256), 256, 0, stream>>>(kw, wtmp, nW);
    switch (tf) {
      case 2: conv_gemm_kernel<2><<<4626, 256, 0, stream>>>(xe_h, wtmp, kb, buf0, L); break;
      case 3: conv_gemm_kernel<3><<<4626, 256, 0, stream>>>(xe_h, wtmp, kb, buf0, L); break;
      case 6: conv_gemm_kernel<6><<<4626, 256, 0, stream>>>(xe_h, wtmp, kb, buf0, L); break;
      default: conv_gemm_kernel<7><<<4626, 256, 0, stream>>>(xe_h, wtmp, kb, buf0, L); break;
    }
    pack_f16_kernel<<<(unsigned)((nW + 255) / 256), 256, 0, stream>>>(qw, wtmp, nW);
    switch (tf) {
      case 2: conv_gemm_kernel<2><<<4626, 256, 0, stream>>>(xe_h, wtmp, qb, buf1, L); break;
      case 3: conv_gemm_kernel<3><<<4626, 256, 0, stream>>>(xe_h, wtmp, qb, buf1, L); break;
      case 6: conv_gemm_kernel<6><<<4626, 256, 0, stream>>>(xe_h, wtmp, qb, buf1, L); break;
      default: conv_gemm_kernel<7><<<4626, 256, 0, stream>>>(xe_h, wtmp, qb, buf1, L); break;
    }

    // residual base then fused attention accumulate
    pack_f16_kernel<<<(64 * 64 + 255) / 256, 256, 0, stream>>>(frw, wtmp, 64 * 64);
    res_gemm_kernel<<<3084, 256, 0, stream>>>(xe_h, wtmp, frb, fob, hout);
    attn_kernel<<<NB * TT, 256, 0, stream>>>(buf0, buf1, vpT, adj, adjh, hout);

    // head BN -> concat (f16)
    zero_f32_kernel<<<1, 256, 0, stream>>>(stats, 128);
    bn_stats_kernel<<<dim3(386, 64), 256, 0, stream>>>(hout, stats);
    bn_apply_concat_kernel<<<(unsigned)((n_xe + 255) / 256), 256, 0, stream>>>(
        hout, stats, bng, bnb, concat, hi);
  }

  // final fused 1x1 (+residual) -> BN -> drop padded node
  pack_finalA_kernel<<<(64 * 320 + 255) / 256, 256, 0, stream>>>(mfo_w, mfr_w, packA);
  final_gemm_kernel<<<3084, 256, 0, stream>>>(concat, xe_h, packA, mfo_b, mfr_b, hout);
  zero_f32_kernel<<<1, 256, 0, stream>>>(stats, 128);
  bn_stats_kernel<<<dim3(386, 64), 256, 0, stream>>>(hout, stats);
  final_out_kernel<<<24576, 256, 0, stream>>>(hout, stats, mbn_g, mbn_b, (float*)d_out);
}
